// _MockAttention_19026705121792
// MI455X (gfx1250) — matrix operand order repaired and checked by reading, not compiled, not run on hardware
//
#include <hip/hip_runtime.h>

// Problem constants (match reference)
#define BATCH   2
#define SEQ     2048
#define DMODEL  1024
#define NHEADS  16
#define HDIM    64              // DMODEL / NHEADS
#define MTOT    (BATCH * SEQ)   // 4096 rows

typedef __attribute__((ext_vector_type(16))) _Float16 v16h;
typedef __attribute__((ext_vector_type(8)))  float    v8f;
typedef __attribute__((ext_vector_type(8)))  unsigned v8u;

// ---------------- CDNA5 async copy to LDS (ASYNCcnt path) ----------------
#if __has_builtin(__builtin_amdgcn_global_load_async_to_lds_b128)
#define HAVE_ASYNC_LDS 1
typedef int v4i_ __attribute__((vector_size(16)));
typedef __attribute__((address_space(1))) v4i_ g_v4i;   // global int4
typedef __attribute__((address_space(3))) v4i_ l_v4i;   // LDS int4
#else
#define HAVE_ASYNC_LDS 0
#endif

static __device__ __forceinline__ void cp16_lds(_Float16* dst_lds,
                                                const _Float16* src_g) {
#if HAVE_ASYNC_LDS
  _Float16* sg = const_cast<_Float16*>(src_g);
  __builtin_amdgcn_global_load_async_to_lds_b128(
      (g_v4i*)sg, (l_v4i*)dst_lds, 0, 0);
#else
  *(uint4*)dst_lds = *(const uint4*)src_g;
#endif
}

template <int N>
static __device__ __forceinline__ void wait_async() {
#if HAVE_ASYNC_LDS
#if __has_builtin(__builtin_amdgcn_s_wait_asynccnt)
  __builtin_amdgcn_s_wait_asynccnt((unsigned short)N);
#else
  asm volatile("s_wait_asynccnt %0" ::"i"(N) : "memory");
#endif
#endif
}

// ---------------- WMMA fragment packing (ISA 7.12.2, wave32, 16-bit) -----
static __device__ __forceinline__ v16h as_v16h(v8u u) {
  union { v8u u; v16h h; } c; c.u = u; return c.h;
}

// A 16x32 (MxK): lane holds row M=(L&15); pair sets {0..3,8..11} / {4..7,12..15}
static __device__ __forceinline__ v16h pack_a_frag(const unsigned* rp) {
  const int hi4 = ((threadIdx.x & 31) >> 4) << 2;
  v8u u;
  u[0] = rp[hi4 + 0]; u[1] = rp[hi4 + 1]; u[2] = rp[hi4 + 2]; u[3] = rp[hi4 + 3];
  u[4] = rp[8 + hi4 + 0]; u[5] = rp[8 + hi4 + 1]; u[6] = rp[8 + hi4 + 2]; u[7] = rp[8 + hi4 + 3];
  return as_v16h(u);
}

// B 32x16 (KxN): lane holds col N=(L&15); pair sets {0..3,8..11} / {4..7,12..15}
static __device__ __forceinline__ v16h pack_b_frag(const unsigned* cp) {
  const int hi8 = ((threadIdx.x & 31) >> 4) << 3;
  v8u u;
#pragma unroll
  for (int v = 0; v < 8; ++v) u[v] = cp[(hi8 >> 1) + (v & 3) + ((v >> 2) << 3)];
  return as_v16h(u);
}

static __device__ __forceinline__ v8f wmma16(v16h a, v16h b, v8f c) {
  return __builtin_amdgcn_wmma_f32_16x16x32_f16(false, a, false, b, (short)0, c,
                                                false, false);
}

// ---------------------------------------------------------------------
__global__ void f32_to_f16(const float* __restrict__ in, _Float16* __restrict__ out,
                           int n) {
  int i = blockIdx.x * blockDim.x + threadIdx.x;
  if (i < n) out[i] = (_Float16)in[i];
}

// ---------------------------------------------------------------------
// GEMM: out[M,N] = A[M,K] @ W[N,K]^T + bias[N]
// 128 threads (4 waves, 2x2), block tile 64x128, wave tile 32x64 (2x4 WMMA),
// K staged 32 at a time through double-buffered LDS via async-to-LDS copies.
// ---------------------------------------------------------------------
template <typename OutT>
__global__ __launch_bounds__(128)
void gemm_bias(const _Float16* __restrict__ A, const _Float16* __restrict__ W,
               const float* __restrict__ bias, OutT* __restrict__ out,
               int M, int K, int N) {
  __shared__ __align__(16) _Float16 sA[2][64 * 32];    // 2 x 4 KB
  __shared__ __align__(16) _Float16 sW[2][128 * 32];   // 2 x 8 KB

  const int tid  = threadIdx.x;
  const int lane = tid & 31;
  const int l15  = lane & 15;
  const int hi   = lane >> 4;
  const int wave = tid >> 5;
  const int wr   = (wave >> 1) << 5;   // 0 / 32
  const int wc   = (wave & 1) << 6;    // 0 / 64
  const int m0   = blockIdx.y * 64;
  const int n0   = blockIdx.x * 128;

  v8f acc[2][4] = {};

  auto stage = [&](int b, int kk) {
#pragma unroll
    for (int i = 0; i < 2; ++i) {                   // A: 64x32 halves
      int e = (tid + i * 128) * 8, r = e >> 5, c = e & 31;
      cp16_lds(&sA[b][e], &A[(size_t)(m0 + r) * K + kk + c]);
    }
#pragma unroll
    for (int i = 0; i < 4; ++i) {                   // W: 128x32 halves
      int e = (tid + i * 128) * 8, r = e >> 5, c = e & 31;
      cp16_lds(&sW[b][e], &W[(size_t)(n0 + r) * K + kk + c]);
    }
  };

  stage(0, 0);
  int buf = 0;
  for (int kk = 0; kk < K; kk += 32) {
    if (kk + 32 < K) { stage(buf ^ 1, kk + 32); wait_async<6>(); }
    else             { wait_async<0>(); }
    __syncthreads();

    const _Float16* pa = sA[buf];
    const _Float16* pw = sW[buf];
    v16h a0 = pack_a_frag((const unsigned*)&pa[(wr + l15) * 32]);
    v16h a1 = pack_a_frag((const unsigned*)&pa[(wr + 16 + l15) * 32]);
    v16h bf[4];
#pragma unroll
    for (int j = 0; j < 4; ++j)
      bf[j] = pack_b_frag((const unsigned*)&pw[(wc + j * 16 + l15) * 32]);
#pragma unroll
    for (int j = 0; j < 4; ++j) {
      acc[0][j] = wmma16(a0, bf[j], acc[0][j]);
      acc[1][j] = wmma16(a1, bf[j], acc[1][j]);
    }
    __syncthreads();       // all waves done reading buf before it is re-staged
    buf ^= 1;
  }

  // epilogue: C layout -> lane holds col l15, rows v + 8*hi
#pragma unroll
  for (int mi = 0; mi < 2; ++mi) {
#pragma unroll
    for (int j = 0; j < 4; ++j) {
      int col = n0 + wc + j * 16 + l15;
      float bv = bias[col];
#pragma unroll
      for (int v = 0; v < 8; ++v) {
        int row = m0 + wr + mi * 16 + v + 8 * hi;
        out[(size_t)row * N + col] = (OutT)(acc[mi][j][v] + bv);
      }
    }
  }
}

// ---------------------------------------------------------------------
// Flash attention.  qkv: [B*SEQ, 3*DMODEL] f16 ([3][H][hd] on last dim).
// grid (SEQ/64, NHEADS, BATCH), block 128 (4 waves); wave owns 16 q rows.
// Row sums of P are accumulated by an extra WMMA against an all-ones tile
// (l = P @ 1), removing the per-row sum butterflies entirely.
// ---------------------------------------------------------------------
__global__ __launch_bounds__(128)
void attention_kernel(const _Float16* __restrict__ qkv,
                      _Float16* __restrict__ attn_out) {
  __shared__ __align__(16) _Float16 sVt[HDIM * 32];   // V^T tile [d][key]
  __shared__ __align__(16) _Float16 sP[4][16 * 32];   // per-wave P tile [q][key]

  const int tid  = threadIdx.x;
  const int lane = tid & 31;
  const int l15  = lane & 15;
  const int hi   = lane >> 4;
  const int wave = tid >> 5;

  const int qt = blockIdx.x;
  const int h  = blockIdx.y;
  const int b  = blockIdx.z;
  const int qbase = qt * 64 + wave * 16;

  const size_t rstride = 3 * (size_t)DMODEL;
  const _Float16* Qb = qkv + (size_t)b * SEQ * rstride + h * HDIM;
  const _Float16* Kb = Qb + DMODEL;
  const _Float16* Vb = Qb + 2 * DMODEL;

  const _Float16* qrow = Qb + (size_t)(qbase + l15) * rstride;
  const v16h qf0 = pack_a_frag((const unsigned*)qrow);
  const v16h qf1 = pack_a_frag((const unsigned*)(qrow + 32));

  v16h vone;
#pragma unroll
  for (int i = 0; i < 16; ++i) vone[i] = (_Float16)1.0f;

  float mrow[8];
  v8f o[4] = {};
  v8f lacc = {};
#pragma unroll
  for (int v = 0; v < 8; ++v) mrow[v] = -1e30f;
  const float scale = 0.125f;   // hd^-0.5

  for (int j0 = 0; j0 < SEQ; j0 += 32) {
    // stage V[j0..j0+31][0..63] transposed into sVt[d][key]
#pragma unroll
    for (int i = 0; i < 2; ++i) {
      int e = (tid + i * 128) * 8;
      int key = e >> 6, d = e & 63;
      uint4 val = *(const uint4*)(Vb + (size_t)(j0 + key) * rstride + d);
      const _Float16* hs = (const _Float16*)&val;
#pragma unroll
      for (int x = 0; x < 8; ++x) sVt[(d + x) * 32 + key] = hs[x];
    }

    // S = Q K^T for two 16-key subtiles
    const _Float16* krow0 = Kb + (size_t)(j0 + l15) * rstride;
    const _Float16* krow1 = Kb + (size_t)(j0 + 16 + l15) * rstride;
    v8f s0 = {}, s1 = {};
    s0 = wmma16(qf0, pack_b_frag((const unsigned*)krow0), s0);
    s0 = wmma16(qf1, pack_b_frag((const unsigned*)(krow0 + 32)), s0);
    s1 = wmma16(qf0, pack_b_frag((const unsigned*)krow1), s1);
    s1 = wmma16(qf1, pack_b_frag((const unsigned*)(krow1 + 32)), s1);

    // online softmax; row r = v + 8*hi lives across the 16 lanes of this half
#pragma unroll
    for (int v = 0; v < 8; ++v) {
      float a = s0[v] * scale, c = s1[v] * scale;
      float mx = fmaxf(a, c);
#pragma unroll
      for (int m = 1; m <= 8; m <<= 1) mx = fmaxf(mx, __shfl_xor(mx, m, 32));
      float mnew  = fmaxf(mrow[v], mx);
      float alpha = __expf(mrow[v] - mnew);
      mrow[v] = mnew;
      float p0 = __expf(a - mnew);
      float p1 = __expf(c - mnew);
      lacc[v] *= alpha;
#pragma unroll
      for (int t = 0; t < 4; ++t) o[t][v] *= alpha;
      int pr = (v + 8 * hi) * 32;
      sP[wave][pr + l15]      = (_Float16)p0;
      sP[wave][pr + 16 + l15] = (_Float16)p1;
    }
    __syncthreads();   // sVt + sP visible

    // O += P V, and l += P @ ones (row sums broadcast into every lane)
    v16h pf = pack_a_frag((const unsigned*)&sP[wave][l15 * 32]);
#pragma unroll
    for (int t = 0; t < 4; ++t) {
      v16h vf = pack_b_frag((const unsigned*)&sVt[(t * 16 + l15) * 32]);
      o[t] = wmma16(pf, vf, o[t]);
    }
    lacc = wmma16(pf, vone, lacc);
    __syncthreads();   // protect sVt/sP before next iteration overwrites
  }

  // normalize and write [B*SEQ, DMODEL] f16
#pragma unroll
  for (int v = 0; v < 8; ++v) {
    float inv = 1.0f / lacc[v];
    int r = qbase + v + 8 * hi;
    size_t orow = ((size_t)b * SEQ + r) * DMODEL + h * HDIM;
#pragma unroll
    for (int t = 0; t < 4; ++t)
      attn_out[orow + t * 16 + l15] = (_Float16)(o[t][v] * inv);
  }
}

// ---------------------------------------------------------------------
extern "C" void kernel_launch(void* const* d_in, const int* in_sizes, int n_in,
                              void* d_out, int out_size, void* d_ws, size_t ws_size,
                              hipStream_t stream) {
  const float* x      = (const float*)d_in[0];
  const float* qkv_w  = (const float*)d_in[1];
  const float* qkv_b  = (const float*)d_in[2];
  const float* proj_w = (const float*)d_in[3];
  const float* proj_b = (const float*)d_in[4];
  float* out = (float*)d_out;

  char* ws = (char*)d_ws;
  _Float16* x16     = (_Float16*)ws;  ws += (size_t)MTOT * DMODEL * 2;        // 8 MB
  _Float16* qkvw16  = (_Float16*)ws;  ws += (size_t)3 * DMODEL * DMODEL * 2;  // 6 MB
  _Float16* projw16 = (_Float16*)ws;  ws += (size_t)DMODEL * DMODEL * 2;      // 2 MB
  _Float16* qkv16   = (_Float16*)ws;  ws += (size_t)MTOT * 3 * DMODEL * 2;    // 24 MB
  _Float16* attn16  = (_Float16*)ws;                                          // 8 MB

  const int nx = MTOT * DMODEL;
  const int nw = 3 * DMODEL * DMODEL;
  const int np = DMODEL * DMODEL;
  f32_to_f16<<<(nx + 255) / 256, 256, 0, stream>>>(x, x16, nx);
  f32_to_f16<<<(nw + 255) / 256, 256, 0, stream>>>(qkv_w, qkvw16, nw);
  f32_to_f16<<<(np + 255) / 256, 256, 0, stream>>>(proj_w, projw16, np);

  // qkv = x @ qkv_w^T + qkv_b   [4096, 3072] f16
  gemm_bias<_Float16><<<dim3(3 * DMODEL / 128, MTOT / 64), 128, 0, stream>>>(
      x16, qkvw16, qkv_b, qkv16, MTOT, DMODEL, 3 * DMODEL);

  // flash attention -> attn16 [4096, 1024] f16
  attention_kernel<<<dim3(SEQ / 64, NHEADS, BATCH), 128, 0, stream>>>(qkv16, attn16);

  // out = attn @ proj_w^T + proj_b  [4096, 1024] f32
  gemm_bias<float><<<dim3(DMODEL / 128, MTOT / 64), 128, 0, stream>>>(
      attn16, projw16, proj_b, out, MTOT, DMODEL, DMODEL);
}
